// LagunaAttention_36369783062550
// MI455X (gfx1250) — matrix operand order repaired and checked by reading, not compiled, not run on hardware
//
#include <hip/hip_runtime.h>

#define T_LEN   2048
#define HIDDEN  2048
#define N_HEADS 32
#define N_KV    8
#define HEAD_D  128
#define QSIZE   (N_HEADS * HEAD_D)   // 4096
#define KVSIZE  (N_KV * HEAD_D)      // 1024
#define QKVW    (QSIZE + 2 * KVSIZE) // 6144
#define WINDOW  512

typedef __attribute__((ext_vector_type(16))) __bf16 v16bf;
typedef __attribute__((ext_vector_type(8)))  float  v8f;

// native bf16 convert (RNE) -- single hw instruction on gfx1250
__device__ __forceinline__ unsigned short f2bf(float x) {
    union { __bf16 b; unsigned short u; } c;
    c.b = (__bf16)x;
    return c.u;
}
__device__ __forceinline__ unsigned f2bf2(float lo, float hi) {
    return (unsigned)f2bf(lo) | ((unsigned)f2bf(hi) << 16);
}

// ---- DPP16 cross-lane sum within each 16-lane half (pure VALU) -------------
template<int CTRL>
__device__ __forceinline__ float dppf(float x) {
    union { float f; int i; } a, r;
    a.f = x;
    r.i = __builtin_amdgcn_update_dpp(a.i, a.i, CTRL, 0xF, 0xF, false);
    return r.f;
}
__device__ __forceinline__ float red16_sum(float x) {
    x += dppf<0xB1>(x);    // quad_perm [1,0,3,2]  (xor 1)
    x += dppf<0x4E>(x);    // quad_perm [2,3,0,1]  (xor 2)
    x += dppf<0x141>(x);   // row_half_mirror      (xor 4..7)
    x += dppf<0x140>(x);   // row_mirror           (xor 8..15)
    return x;
}

// A-fragment (16x32 bf16, M x K): lane<16 holds K {0..7,16..23}, lane>=16 holds {8..15,24..31}
__device__ __forceinline__ v16bf frag_a(const unsigned short* tile, int stride, int lane) {
    int r = lane & 15, h = (lane >> 4) & 1;
    union { v16bf v; uint4 q[2]; } u;
    const unsigned short* p = tile + (size_t)r * stride;
    u.q[0] = *(const uint4*)(p + h * 8);
    u.q[1] = *(const uint4*)(p + 16 + h * 8);
    return u.v;
}

// B-fragment (32x16 bf16, K x N) from tile stored [N][K]: lane<16 holds K {0..7,16..23}, lane>=16 holds {8..15,24..31}
__device__ __forceinline__ v16bf frag_b(const unsigned short* tile, int stride, int lane) {
    int c = lane & 15, h = (lane >> 4) & 1;
    union { v16bf v; uint4 q[2]; } u;
    const unsigned short* p = tile + (size_t)c * stride + h * 8;
    u.q[0] = *(const uint4*)(p);
    u.q[1] = *(const uint4*)(p + 16);
    return u.v;
}

__device__ __forceinline__ unsigned lds_off(const void* p) {
    // addrspace(3) -> generic keeps the LDS byte offset in the low 32 bits
    return (unsigned)(size_t)p;
}

// ---------------------------------------------------------------------------
// GEMM: C[M,N] = A[M,K] @ B[K,N], bf16 WMMA with f32 accumulation.
// Block tile 128x128, K-step 32. 8 waves in 2(M) x 4(N); 64x32 per wave.
// A_BF16 path stages the A tile with GLOBAL_LOAD_ASYNC_TO_LDS_B128 (ASYNCcnt),
// overlapped with the f32->bf16 conversion of the B tile. f32 staging uses
// float2 loads + packed-bf16 b32 LDS stores.
// ---------------------------------------------------------------------------
template<bool A_BF16>
__global__ __launch_bounds__(256) void gemm_kernel(
    const void* __restrict__ Ap, const float* __restrict__ B,
    float* __restrict__ C, int M, int N, int K)
{
    __shared__ alignas(16) unsigned short lA[128 * 40]; // [m][k], stride 40
    __shared__ alignas(16) unsigned short lB[128 * 40]; // [n][k], stride 40
    const float* Af = (const float*)Ap;
    const unsigned short* Ab = (const unsigned short*)Ap;
    int tid = threadIdx.x, lane = tid & 31, wave = tid >> 5;
    int wm = wave >> 2, wn = wave & 3;
    int mb = blockIdx.x * 128, nb = blockIdx.y * 128;
    v8f acc[4][2] = {};

    for (int kb = 0; kb < K; kb += 32) {
        __syncthreads();
        if constexpr (A_BF16) {
            // async DMA: 128 rows x 32 bf16 = 512 x 16B chunks straight to LDS
#pragma unroll
            for (int i = 0; i < 2; ++i) {
                int flat = tid + i * 256;
                int rr = flat >> 2;
                int c8 = (flat & 3) * 8;
                unsigned la = lds_off(lA + rr * 40 + c8);
                unsigned go = ((unsigned)(mb + rr) * (unsigned)K + (unsigned)(kb + c8)) * 2u;
                asm volatile("global_load_async_to_lds_b128 %0, %1, %2"
                             :: "v"(la), "v"(go), "s"(Ab) : "memory");
            }
        } else {
            // A tile: 128 rows x 32 k; float2 loads, packed bf16 b32 stores
#pragma unroll
            for (int i = 0; i < 8; ++i) {
                int flat = tid + i * 256;        // 2048 pairs
                int rr = flat >> 4, k = (flat & 15) * 2;
                float2 v = *(const float2*)(Af + (size_t)(mb + rr) * K + kb + k);
                *(unsigned*)(lA + rr * 40 + k) = f2bf2(v.x, v.y);
            }
        }
        // B tile: 32 k x 128 n -> [n][k]; k-pairs, packed b32 stores
#pragma unroll
        for (int i = 0; i < 8; ++i) {
            int flat = tid + i * 256;            // 2048 pairs
            int n = flat & 127, k = (flat >> 7) * 2;
            float b0 = B[(size_t)(kb + k) * N + nb + n];
            float b1 = B[(size_t)(kb + k + 1) * N + nb + n];
            *(unsigned*)(lB + n * 40 + k) = f2bf2(b0, b1);
        }
        if constexpr (A_BF16)
            asm volatile("s_wait_asynccnt 0x0" ::: "memory");
        __syncthreads();
        v16bf afr[4], bfr[2];
#pragma unroll
        for (int i = 0; i < 4; ++i) afr[i] = frag_a(lA + (wm * 64 + i * 16) * 40, 40, lane);
#pragma unroll
        for (int j = 0; j < 2; ++j) bfr[j] = frag_b(lB + (wn * 32 + j * 16) * 40, 40, lane);
#pragma unroll
        for (int i = 0; i < 4; ++i)
#pragma unroll
            for (int j = 0; j < 2; ++j)
                acc[i][j] = __builtin_amdgcn_wmma_f32_16x16x32_bf16(
                    false, afr[i], false, bfr[j], (short)0, acc[i][j], false, false);
    }

    int r = lane & 15, h = (lane >> 4) & 1;
#pragma unroll
    for (int i = 0; i < 4; ++i)
#pragma unroll
        for (int j = 0; j < 2; ++j)
#pragma unroll
            for (int e = 0; e < 8; ++e)
                C[(size_t)(mb + wm * 64 + i * 16 + h * 8 + e) * N +
                  nb + wn * 32 + j * 16 + r] = acc[i][j][e];
}

// ---------------------------------------------------------------------------
// Gate: softplus(h @ w_g). Lanes of a wave share the hidden row (broadcast),
// coalesce on w_g.
// ---------------------------------------------------------------------------
__global__ __launch_bounds__(256) void gate_kernel(
    const float* __restrict__ h, const float* __restrict__ w_g, float* __restrict__ gate)
{
    int idx = blockIdx.x * 256 + threadIdx.x;      // T*32
    int t = idx >> 5, hh = idx & 31;
    const float* hr = h + (size_t)t * HIDDEN;
    float acc = 0.f;
    for (int k = 0; k < HIDDEN; ++k) acc += hr[k] * w_g[k * N_HEADS + hh];
    gate[idx] = fmaxf(acc, 0.f) + log1pf(__expf(-fabsf(acc)));
}

// ---------------------------------------------------------------------------
// RMSNorm + RoPE on q/k, v pass-through; fp32 -> bf16 for the attention stage.
// grid = (T, 48): z<32 -> q head, 32..39 -> k head, 40..47 -> v head copy.
// The 1/sqrt(HEAD_D) attention scale is folded into Q here (commutes with the
// rotation), removing per-element muls from the attention inner loop.
// ---------------------------------------------------------------------------
__global__ __launch_bounds__(128) void norm_rope_kernel(
    const float* __restrict__ qkv, const int* __restrict__ positions,
    const float* __restrict__ qw, const float* __restrict__ kw,
    unsigned short* __restrict__ qbf, unsigned short* __restrict__ kbf,
    unsigned short* __restrict__ vbf)
{
    int t = blockIdx.x, z = blockIdx.y, d = threadIdx.x;
    if (z >= 40) {
        int hh = z - 40;
        float x = qkv[(size_t)t * QKVW + QSIZE + KVSIZE + hh * HEAD_D + d];
        vbf[((size_t)t * N_KV + hh) * HEAD_D + d] = f2bf(x);
        return;
    }
    bool isq = z < 32;
    const float* src = qkv + (size_t)t * QKVW + (isq ? z * HEAD_D : QSIZE + (z - 32) * HEAD_D);
    float x = src[d];
    float ss = x * x;
#pragma unroll
    for (int o = 16; o >= 1; o >>= 1) ss += __shfl_xor(ss, o, 32);
    __shared__ float red[4];
    __shared__ float xs[HEAD_D];
    int wave = d >> 5, lane = d & 31;
    if (lane == 0) red[wave] = ss;
    __syncthreads();
    float var = (red[0] + red[1] + red[2] + red[3]) * (1.0f / HEAD_D);
    float w = (isq ? qw : kw)[d];
    float xn = x * rsqrtf(var + 1e-6f) * w;
    xs[d] = xn;
    __syncthreads();
    int d2 = d & 63;
    float freq = __powf(10000.0f, -(float)d2 * (1.0f / 64.0f));
    float ang = (float)positions[t] * freq;
    float c = cosf(ang), s = sinf(ang);
    float x1 = xs[d2], x2 = xs[d2 + 64];
    float o = (d < 64) ? (x1 * c - x2 * s) : (x2 * c + x1 * s);
    if (isq) {
        o *= 0.08838834764831845f;   // fold 1/sqrt(128) into Q
        qbf[((size_t)t * N_HEADS + z) * HEAD_D + d] = f2bf(o);
    } else {
        kbf[((size_t)t * N_KV + (z - 32)) * HEAD_D + d] = f2bf(o);
    }
}

// ---------------------------------------------------------------------------
// Sliding-window attention, streaming softmax WITHOUT a running max:
// q,k are RMS-normalized, so |score| <= sqrt(HEAD_D) ~= 11.32 by
// Cauchy-Schwarz -> exp(s) <= 8.2e4, window sum <= 4.3e7: safely in fp32.
// The shift m=0 is exact for softmax, so no per-step row-max reduction and no
// accumulator rescaling are needed; the denominator is a per-lane partial sum
// reduced once (DPP) in the epilogue.
// Block = (q-tile of 16, kv-head); 4 waves = 4 q-heads of the GQA group.
// K tile staged with GLOBAL_LOAD_ASYNC_TO_LDS_B128 (ASYNCcnt).
// ---------------------------------------------------------------------------
__global__ __launch_bounds__(128) void attn_kernel(
    const unsigned short* __restrict__ qbf, const unsigned short* __restrict__ kbf,
    const unsigned short* __restrict__ vbf, const float* __restrict__ gate,
    unsigned short* __restrict__ ag)
{
    __shared__ alignas(16) unsigned short Kt[32 * 136];   // [key][d]
    __shared__ alignas(16) unsigned short Vt[128 * 40];   // [d][key] (transposed)
    __shared__ alignas(16) unsigned short Pt[4][16 * 40]; // per-wave P [qrow][key]

    int tid = threadIdx.x, lane = tid & 31, wave = tid >> 5;
    int qt = blockIdx.x, kv = blockIdx.y;
    int head = kv * 4 + wave;
    int qb = qt * 16;
    int r = lane & 15, h = (lane >> 4) & 1;

    v16bf qf[4];
    {
        const unsigned short* qbase = qbf + ((size_t)qb * N_HEADS + head) * HEAD_D;
#pragma unroll
        for (int c = 0; c < 4; ++c) qf[c] = frag_a(qbase + c * 32, N_HEADS * HEAD_D, lane);
    }
    v8f out[8] = {};
    float lsum[8];
#pragma unroll
    for (int e = 0; e < 8; ++e) lsum[e] = 0.f;

    int kt_lo = qt > 32 ? qt - 32 : 0;
    for (int kt = kt_lo; kt <= qt; kt += 2) {
        int ks = kt * 16;
        __syncthreads();
        // --- K tile: async DMA global->LDS, 32 rows x 256B in 16B chunks ----
#pragma unroll
        for (int i = 0; i < 4; ++i) {
            int flat = tid + i * 128;            // 512 chunks
            int key = flat >> 4;
            int c8 = (flat & 15) * 8;            // element offset in row
            if (ks + key < T_LEN) {
                unsigned la = lds_off(Kt + key * 136 + c8);
                unsigned go = (((unsigned)(ks + key) * N_KV + kv) * HEAD_D + c8) * 2u;
                asm volatile("global_load_async_to_lds_b128 %0, %1, %2"
                             :: "v"(la), "v"(go), "s"(kbf) : "memory");
            } else {
                *(uint4*)(Kt + key * 136 + c8) = make_uint4(0u, 0u, 0u, 0u);
            }
        }
        // --- V tile: load + transpose into [d][key] (overlaps the async K) --
        for (int i = 0; i < 8; ++i) {
            int flat = tid + i * 128;
            int key = flat >> 5;
            int dp = (flat & 31) * 4;
            uint2 vval = make_uint2(0u, 0u);
            if (ks + key < T_LEN)
                vval = *(const uint2*)(vbf + (((size_t)(ks + key)) * N_KV + kv) * HEAD_D + dp);
            unsigned short* vp = (unsigned short*)&vval;
#pragma unroll
            for (int j = 0; j < 4; ++j) Vt[(dp + j) * 40 + key] = vp[j];
        }
        asm volatile("s_wait_asynccnt 0x0" ::: "memory");
        __syncthreads();

        // S = Q @ K^T (Q pre-scaled), contraction over d=128; 2 key subtiles
        v8f s0 = {}, s1 = {};
#pragma unroll
        for (int c = 0; c < 4; ++c) {
            v16bf k0 = frag_b(Kt + c * 32, 136, lane);
            v16bf k1 = frag_b(Kt + 16 * 136 + c * 32, 136, lane);
            s0 = __builtin_amdgcn_wmma_f32_16x16x32_bf16(false, qf[c], false, k0, (short)0, s0, false, false);
            s1 = __builtin_amdgcn_wmma_f32_16x16x32_bf16(false, qf[c], false, k1, (short)0, s1, false, false);
        }

        // p = mask ? exp(s) : 0   (no max shift needed; scores are bounded)
        unsigned short* P = Pt[wave];
#pragma unroll
        for (int e = 0; e < 8; ++e) {
            int qpos = qb + h * 8 + e;
            int d0 = qpos - (ks + r);
            int d1 = qpos - (ks + 16 + r);
            float p0 = (d0 >= 0 && d0 < WINDOW) ? __expf(s0[e]) : 0.f;
            float p1 = (d1 >= 0 && d1 < WINDOW) ? __expf(s1[e]) : 0.f;
            lsum[e] += p0 + p1;
            P[(h * 8 + e) * 40 + r] = f2bf(p0);
            P[(h * 8 + e) * 40 + 16 + r] = f2bf(p1);
        }

        // out += P @ V  (A = P 16x32 keys, B = V [key][d] staged transposed)
        v16bf pf = frag_a(P, 40, lane);
#pragma unroll
        for (int n = 0; n < 8; ++n) {
            v16bf vf = frag_b(Vt + n * 16 * 40, 40, lane);
            out[n] = __builtin_amdgcn_wmma_f32_16x16x32_bf16(false, pf, false, vf, (short)0, out[n], false, false);
        }
    }

    // epilogue: single cross-lane denominator reduction, /l, * gate, bf16 store
#pragma unroll
    for (int e = 0; e < 8; ++e) {
        int row = qb + h * 8 + e;
        float l = red16_sum(lsum[e]);
        float g = gate[row * N_HEADS + head] / l;
#pragma unroll
        for (int n = 0; n < 8; ++n)
            ag[(size_t)row * QSIZE + head * HEAD_D + n * 16 + r] = f2bf(out[n][e] * g);
    }
}

// ---------------------------------------------------------------------------
extern "C" void kernel_launch(void* const* d_in, const int* in_sizes, int n_in,
                              void* d_out, int out_size, void* d_ws, size_t ws_size,
                              hipStream_t stream)
{
    (void)in_sizes; (void)n_in; (void)out_size; (void)ws_size;
    const int*   positions = (const int*)d_in[0];
    const float* hidden    = (const float*)d_in[1];
    const float* w_qkv     = (const float*)d_in[2];
    const float* w_o       = (const float*)d_in[3];
    const float* w_g       = (const float*)d_in[4];
    const float* q_norm_w  = (const float*)d_in[5];
    const float* k_norm_w  = (const float*)d_in[6];

    char* ws = (char*)d_ws;
    float*          qkv  = (float*)(ws);                       // 2048*6144 f32  = 48 MB
    unsigned short* qbf  = (unsigned short*)(ws + 50331648);   // 2048*4096 bf16 = 16 MB
    unsigned short* kbf  = (unsigned short*)(ws + 67108864);   // 2048*1024 bf16 =  4 MB
    unsigned short* vbf  = (unsigned short*)(ws + 71303168);   // 2048*1024 bf16 =  4 MB
    float*          gate = (float*)(ws + 75497472);            // 2048*32 f32
    unsigned short* ag   = (unsigned short*)(ws + 75759616);   // 2048*4096 bf16 = 16 MB

    // 1) qkv = hidden @ w_qkv   (f32 in, bf16 WMMA, f32 out)
    gemm_kernel<false><<<dim3(T_LEN / 128, QKVW / 128), 256, 0, stream>>>(
        (const void*)hidden, w_qkv, qkv, T_LEN, QKVW, HIDDEN);

    // 2) gate = softplus(hidden @ w_g)
    gate_kernel<<<dim3((T_LEN * N_HEADS) / 256), 256, 0, stream>>>(hidden, w_g, gate);

    // 3) RMSNorm + RoPE -> bf16 q/k (Q pre-scaled), bf16 v
    norm_rope_kernel<<<dim3(T_LEN, 48), 128, 0, stream>>>(
        qkv, positions, q_norm_w, k_norm_w, qbf, kbf, vbf);

    // 4) sliding-window attention (+gate) -> bf16 attn
    attn_kernel<<<dim3(T_LEN / 16, N_KV), 128, 0, stream>>>(qbf, kbf, vbf, gate, ag);

    // 5) out = attn_gated @ w_o  (bf16 A staged via async DMA)
    gemm_kernel<true><<<dim3(T_LEN / 128, HIDDEN / 128), 256, 0, stream>>>(
        (const void*)ag, w_o, (float*)d_out, T_LEN, HIDDEN, QSIZE);
}
